// InternAttention_36979668418942
// MI455X (gfx1250) — compile-verified, matrix operand order checked by reading, not run on hardware
//
#include <hip/hip_runtime.h>
#include <hip/hip_bf16.h>

// ---------------- problem constants ----------------
constexpr int B    = 8;
constexpr int N    = 1025;
constexpr int NPAD = 1040;          // 65 tiles of 16
constexpr int C    = 3200;
constexpr int H    = 25;
constexpr int D    = 128;
constexpr int O3   = 3 * C;         // 9600
constexpr int MROWS = B * NPAD;     // 8320 = 65 * 128

typedef __bf16 bf16;
typedef __attribute__((ext_vector_type(16))) bf16  v16bf;
typedef __attribute__((ext_vector_type(8)))  bf16  v8bf;
typedef __attribute__((ext_vector_type(8)))  float v8f;

// ---------------- WMMA helpers ----------------
// A-operand (16x32 bf16, row-major source) per ISA 7.12.2:
//   lane<16 : row = lane,     K = {k0+0..7,  k0+16..23}
//   lane>=16: row = lane-16,  K = {k0+8..15, k0+24..31}
// B-operand (Kx16) stored as 16 rows x K (B^T row-major) has the identical
// per-lane pattern, so one loader serves both.
__device__ __forceinline__ v16bf load_frag(const bf16* __restrict__ base, int ld,
                                           int row, int k0, int lane) {
  int r    = row + (lane & 15);
  int koff = (lane & 16) ? 8 : 0;
  const bf16* p = base + (size_t)r * ld + (k0 + koff);
  v8bf lo = *(const v8bf*)(p);
  v8bf hi = *(const v8bf*)(p + 16);
  v16bf f;
#pragma unroll
  for (int i = 0; i < 8; ++i) { f[i] = lo[i]; f[i + 8] = hi[i]; }
  return f;
}

__device__ __forceinline__ v8f wmma_bf16(v16bf a, v16bf b, v8f c) {
  return __builtin_amdgcn_wmma_f32_16x16x32_bf16(false, a, false, b,
                                                 (short)0, c, false, false);
}

// Async global->LDS copy of one 16-byte chunk (per-lane), tracked by ASYNCcnt.
// LDS aperture lives in flat-addr bits [63:32] (ISA 10.2), so the low 32 bits
// of a flat pointer to __shared__ ARE the DS byte offset.
__device__ __forceinline__ void async_copy16(const bf16* g, bf16* l) {
  unsigned lds_off = (unsigned)(uintptr_t)l;
  unsigned long long ga = (unsigned long long)(uintptr_t)g;
  asm volatile("global_load_async_to_lds_b128 %0, %1, off"
               :: "v"(lds_off), "v"(ga) : "memory");
}

__device__ __forceinline__ void wait_async0() {
  asm volatile("s_wait_asynccnt 0x0" ::: "memory");
}

// ---------------- conversion kernels ----------------
__global__ void cvt_bf16_kernel(const float* __restrict__ src,
                                bf16* __restrict__ dst, size_t n) {
  size_t i = (size_t)blockIdx.x * blockDim.x + threadIdx.x;
  if (i < n) dst[i] = (bf16)src[i];
}

// x (B,N,C) f32 -> xb (B,NPAD,C) bf16 with zero padding for n >= N
__global__ void cvt_pad_x_kernel(const float* __restrict__ x,
                                 bf16* __restrict__ xb) {
  size_t i = (size_t)blockIdx.x * blockDim.x + threadIdx.x;
  size_t total = (size_t)B * NPAD * C;
  if (i >= total) return;
  int c = (int)(i % C);
  int t = (int)(i / C);
  int n = t % NPAD;
  int b = t / NPAD;
  xb[i] = (n < N) ? (bf16)x[((size_t)(b * N + n)) * C + c] : (bf16)0.0f;
}

// ---------------- bf16 WMMA GEMM:  Cout = A(MxK) * Bw(NoutxK)^T ----------------
// Double-buffered async global->LDS staging: per K-step each block stages a
// 128x32 A tile and 128x32 B^T tile (16KB) while computing on the other buffer.
// mode 0: write f32 to Cout with leading dim Nout (qkv buffer)
// mode 1: row r -> (b=r/NPAD, n=r%NPAD); d_out[(b*N+n)*Nout+c] = v + bias[c]
constexpr int LDSLD = 40;   // 32 k-elems + 8 pad (80B rows: 16B-aligned chunks)

__global__ __launch_bounds__(256) void gemm_bf16_kernel(
    const bf16* __restrict__ A, const bf16* __restrict__ Bw,
    float* __restrict__ Cout, const float* __restrict__ bias,
    int K, int Nout, int mode) {
  __shared__ bf16 lds_a[2][128 * LDSLD];
  __shared__ bf16 lds_b[2][128 * LDSLD];

  int tid  = threadIdx.x;
  int lane = tid & 31;
  int wid  = tid >> 5;
  int wm   = wid & 3;            // 4 waves along M
  int wn   = wid >> 2;           // 2 waves along N
  int m0   = blockIdx.y * 128;
  int n0   = blockIdx.x * 128;

  // per-thread staging chunks: tile = 128 rows x 4 chunks of 8 bf16 = 512 chunks
  int ca = tid, cb = tid + 256;
  int ra = ca >> 2, oa = (ca & 3) * 8;
  int rb = cb >> 2, ob = (cb & 3) * 8;

  auto issue = [&](int buf, int k0) {
    async_copy16(A  + (size_t)(m0 + ra) * K + k0 + oa, &lds_a[buf][ra * LDSLD + oa]);
    async_copy16(A  + (size_t)(m0 + rb) * K + k0 + ob, &lds_a[buf][rb * LDSLD + ob]);
    async_copy16(Bw + (size_t)(n0 + ra) * K + k0 + oa, &lds_b[buf][ra * LDSLD + oa]);
    async_copy16(Bw + (size_t)(n0 + rb) * K + k0 + ob, &lds_b[buf][rb * LDSLD + ob]);
  };

  v8f acc[2][4];
#pragma unroll
  for (int t = 0; t < 2; ++t)
#pragma unroll
    for (int j = 0; j < 4; ++j)
#pragma unroll
      for (int i = 0; i < 8; ++i) acc[t][j][i] = 0.0f;

  int nsteps = K / 32;
  issue(0, 0);
  for (int s = 0; s < nsteps; ++s) {
    int cur = s & 1;
    wait_async0();       // my chunks for buffer `cur` have landed
    __syncthreads();     // everyone's chunks visible; prev reads of buf cur^1 done
    if (s + 1 < nsteps) issue(cur ^ 1, (s + 1) * 32);   // overlap with compute

    const bf16* at = lds_a[cur];
    const bf16* bt = lds_b[cur];
    v16bf a0 = load_frag(at, LDSLD, wm * 32,      0, lane);
    v16bf a1 = load_frag(at, LDSLD, wm * 32 + 16, 0, lane);
#pragma unroll
    for (int j = 0; j < 4; ++j) {
      v16bf bf = load_frag(bt, LDSLD, wn * 64 + j * 16, 0, lane);
      acc[0][j] = wmma_bf16(a0, bf, acc[0][j]);
      acc[1][j] = wmma_bf16(a1, bf, acc[1][j]);
    }
  }

  int col   = lane & 15;
  int rbase = (lane & 16) ? 8 : 0;
#pragma unroll
  for (int t = 0; t < 2; ++t)
#pragma unroll
    for (int j = 0; j < 4; ++j)
#pragma unroll
      for (int i = 0; i < 8; ++i) {
        int r = m0 + wm * 32 + t * 16 + rbase + i;
        int c = n0 + wn * 64 + j * 16 + col;
        float v = acc[t][j][i];
        if (mode == 0) {
          Cout[(size_t)r * Nout + c] = v;
        } else {
          int bb = r / NPAD, n = r % NPAD;
          if (n < N) Cout[((size_t)(bb * N + n)) * Nout + c] = v + bias[c];
        }
      }
}

// ---------------- RMSNorm + split/transpose ----------------
// qkv (MROWS, 9600) f32 -> qb,kb [b][h][n][d] bf16 ; vT [b][h][d][n] bf16
__global__ __launch_bounds__(256) void rmsnorm_split_kernel(
    const float* __restrict__ qkv, const float* __restrict__ qw,
    const float* __restrict__ kw, bf16* __restrict__ qb,
    bf16* __restrict__ kb, bf16* __restrict__ vT) {
  int r = blockIdx.x;                 // global padded row
  int b = r / NPAD, n = r % NPAD;
  const float* row = qkv + (size_t)r * O3;
  int tid = threadIdx.x;

  float sq = 0.0f, sk = 0.0f;
  for (int c = tid; c < C; c += 256) {
    float a = row[c];       sq += a * a;
    float e = row[C + c];   sk += e * e;
  }
  __shared__ float s1[256], s2[256];
  s1[tid] = sq; s2[tid] = sk;
  __syncthreads();
  for (int st = 128; st > 0; st >>= 1) {
    if (tid < st) { s1[tid] += s1[tid + st]; s2[tid] += s2[tid + st]; }
    __syncthreads();
  }
  float qs = rsqrtf(s1[0] / (float)C + 1e-6f);
  float ks = rsqrtf(s2[0] / (float)C + 1e-6f);

  for (int c = tid; c < C; c += 256) {
    int h = c / D, d = c % D;
    size_t qi = ((size_t)(b * H + h) * NPAD + n) * D + d;
    qb[qi] = (bf16)(row[c] * qs * qw[c]);
    kb[qi] = (bf16)(row[C + c] * ks * kw[c]);
    vT[((size_t)(b * H + h) * D + d) * NPAD + n] = (bf16)row[2 * C + c];
  }
}

// ---------------- flash attention: one wave per (b, h, 16-query tile) ----------------
__global__ __launch_bounds__(32) void attn_kernel(
    const bf16* __restrict__ Q, const bf16* __restrict__ Km,
    const bf16* __restrict__ Vt, bf16* __restrict__ O) {
  __shared__ bf16 lds_p[16 * 32];     // P tile staging (C-layout -> A-layout transpose)

  int lane = threadIdx.x;
  int b = blockIdx.z, h = blockIdx.y;
  int q0 = blockIdx.x * 16;
  const bf16* qh = Q  + ((size_t)(b * H + h)) * NPAD * D;
  const bf16* kh = Km + ((size_t)(b * H + h)) * NPAD * D;
  const bf16* vh = Vt + ((size_t)(b * H + h)) * D * NPAD;

  v16bf qa[4];
#pragma unroll
  for (int kc = 0; kc < 4; ++kc) qa[kc] = load_frag(qh, D, q0, kc * 32, lane);

  v8f acc[8];
  float mrow[8], lrow[8];
#pragma unroll
  for (int j = 0; j < 8; ++j) {
    mrow[j] = -1e30f; lrow[j] = 0.0f;
#pragma unroll
    for (int i = 0; i < 8; ++i) acc[j][i] = 0.0f;
  }

  const float scale = 0.08838834764831845f;   // 1/sqrt(128)
  int col   = lane & 15;
  int rbase = (lane & 16) ? 8 : 0;

  for (int kb = 0; kb < NPAD; kb += 32) {
    v8f s0, s1;
#pragma unroll
    for (int i = 0; i < 8; ++i) { s0[i] = 0.0f; s1[i] = 0.0f; }
#pragma unroll
    for (int kc = 0; kc < 4; ++kc) {
      s0 = wmma_bf16(qa[kc], load_frag(kh, D, kb,      kc * 32, lane), s0);
      s1 = wmma_bf16(qa[kc], load_frag(kh, D, kb + 16, kc * 32, lane), s1);
    }
    bool val0 = (kb + col) < N;
    bool val1 = (kb + 16 + col) < N;

#pragma unroll
    for (int i = 0; i < 8; ++i) {
      float a0 = val0 ? s0[i] * scale : -1e30f;
      float a1 = val1 ? s1[i] * scale : -1e30f;
      float mx = fmaxf(a0, a1);
#pragma unroll
      for (int off = 8; off >= 1; off >>= 1)
        mx = fmaxf(mx, __shfl_xor(mx, off, 16));
      float mn = fmaxf(mrow[i], mx);
      float sc = __expf(mrow[i] - mn);
      mrow[i] = mn;
      float p0 = __expf(a0 - mn);
      float p1 = __expf(a1 - mn);
      float rs = p0 + p1;
#pragma unroll
      for (int off = 8; off >= 1; off >>= 1)
        rs += __shfl_xor(rs, off, 16);
      lrow[i] = lrow[i] * sc + rs;
#pragma unroll
      for (int j = 0; j < 8; ++j) acc[j][i] *= sc;
      // stage P (bf16) into LDS row-major 16x32
      lds_p[(rbase + i) * 32 + col]      = (bf16)p0;
      lds_p[(rbase + i) * 32 + 16 + col] = (bf16)p1;
    }
    __syncthreads();
    v16bf pf = load_frag(lds_p, 32, 0, 0, lane);
    __syncthreads();

#pragma unroll
    for (int j = 0; j < 8; ++j)
      acc[j] = wmma_bf16(pf, load_frag(vh, NPAD, j * 16, kb, lane), acc[j]);
  }

  // write O [b][n][h*D + d] as bf16 (pad rows written too; never read into d_out)
#pragma unroll
  for (int j = 0; j < 8; ++j) {
#pragma unroll
    for (int i = 0; i < 8; ++i) {
      float inv = 1.0f / lrow[i];
      int qrow = q0 + rbase + i;
      O[((size_t)(b * NPAD + qrow)) * C + h * D + j * 16 + col] =
          (bf16)(acc[j][i] * inv);
    }
  }
}

// ---------------- host launcher ----------------
extern "C" void kernel_launch(void* const* d_in, const int* in_sizes, int n_in,
                              void* d_out, int out_size, void* d_ws, size_t ws_size,
                              hipStream_t stream) {
  const float* x      = (const float*)d_in[0];
  const float* w_qkv  = (const float*)d_in[1];
  const float* qw     = (const float*)d_in[2];
  const float* kw     = (const float*)d_in[3];
  const float* w_proj = (const float*)d_in[4];
  const float* b_proj = (const float*)d_in[5];
  float* out = (float*)d_out;

  char* ws = (char*)d_ws;
  size_t off = 0;
  auto take = [&](size_t bytes) -> char* {
    char* p = ws + off;
    off += (bytes + 255) & ~(size_t)255;
    return p;
  };

  bf16*  xb     = (bf16*)take((size_t)MROWS * C * sizeof(bf16));
  bf16*  wqkvb  = (bf16*)take((size_t)O3 * C * sizeof(bf16));
  bf16*  wprojb = (bf16*)take((size_t)C * C * sizeof(bf16));
  float* qkvf   = (float*)take((size_t)MROWS * O3 * sizeof(float));
  bf16*  qb     = (bf16*)take((size_t)MROWS * C * sizeof(bf16));
  bf16*  kb     = (bf16*)take((size_t)MROWS * C * sizeof(bf16));
  bf16*  vTb    = (bf16*)take((size_t)MROWS * C * sizeof(bf16));
  bf16*  attnb  = (bf16*)take((size_t)MROWS * C * sizeof(bf16));

  // 1) conversions
  {
    size_t nx = (size_t)MROWS * C;
    cvt_pad_x_kernel<<<(unsigned)((nx + 255) / 256), 256, 0, stream>>>(x, xb);
    size_t nw1 = (size_t)O3 * C;
    cvt_bf16_kernel<<<(unsigned)((nw1 + 255) / 256), 256, 0, stream>>>(w_qkv, wqkvb, nw1);
    size_t nw2 = (size_t)C * C;
    cvt_bf16_kernel<<<(unsigned)((nw2 + 255) / 256), 256, 0, stream>>>(w_proj, wprojb, nw2);
  }

  // 2) QKV GEMM: (8320 x 3200) * (9600 x 3200)^T -> f32 (8320 x 9600)
  gemm_bf16_kernel<<<dim3(O3 / 128, MROWS / 128), 256, 0, stream>>>(
      xb, wqkvb, qkvf, nullptr, C, O3, 0);

  // 3) RMSNorm + split + v-transpose
  rmsnorm_split_kernel<<<MROWS, 256, 0, stream>>>(qkvf, qw, kw, qb, kb, vTb);

  // 4) flash attention: grid (65 q-tiles, 25 heads, 8 batches), 1 wave each
  attn_kernel<<<dim3(NPAD / 16, H, B), 32, 0, stream>>>(qb, kb, vTb, attnb);

  // 5) output projection + bias -> d_out (f32), pad rows dropped
  gemm_bf16_kernel<<<dim3(C / 128, MROWS / 128), 256, 0, stream>>>(
      attnb, wprojb, out, b_proj, C, C, 1);
}
